// VirusHostCoexistenceModel_66168266162278
// MI455X (gfx1250) — matrix operand order repaired and checked by reading, not compiled, not run on hardware
//
#include <hip/hip_runtime.h>
#include <hip/hip_bf16.h>
#include <math.h>

// Problem dimensions (fixed by the reference)
#define NVN 6000
#define NHN 6000
#define DVD 1024
#define DHD 1024
#define HID 128
#define HEADS 3
#define EVE 200000
#define EHE 200000
#define ECE 32768
#define FDIM (HEADS * HID) // 384

typedef __attribute__((ext_vector_type(16))) __bf16 bf16x16;
typedef __attribute__((ext_vector_type(8)))  float  floatx8;
typedef __attribute__((ext_vector_type(4)))  unsigned int u32x4;

// exact GCC-vector type the async-LDS builtin expects (per hipcc diagnostic)
typedef int gcc_v4i __attribute__((vector_size(4 * sizeof(int))));
typedef __attribute__((address_space(1))) gcc_v4i as1_v4i;  // global
typedef __attribute__((address_space(3))) gcc_v4i as3_v4i;  // LDS

union FragU { u32x4 q[2]; bf16x16 v; };

#if __has_builtin(__builtin_amdgcn_global_load_async_to_lds_b128)
#define HAVE_ASYNC_LDS 1
#else
#define HAVE_ASYNC_LDS 0
#endif

template <int Ncnt>
__device__ __forceinline__ void wait_asynccnt_le() {
#if __has_builtin(__builtin_amdgcn_s_wait_asynccnt)
  __builtin_amdgcn_s_wait_asynccnt(Ncnt);
#else
  asm volatile("s_wait_asynccnt %0" ::"i"(Ncnt) : "memory");
#endif
}

// ---------------------------------------------------------------------------
// conversion kernels
// ---------------------------------------------------------------------------
__global__ void k_cvt(const float* __restrict__ x, __bf16* __restrict__ y, long n) {
  long i = (long)blockIdx.x * blockDim.x + threadIdx.x;
  if (i < n) y[i] = (__bf16)x[i];
}

// x: [R,C] row-major f32  ->  y: [C,R] row-major bf16
__global__ void k_cvtT(const float* __restrict__ x, __bf16* __restrict__ y, int R, int C) {
  long i = (long)blockIdx.x * blockDim.x + threadIdx.x;
  long n = (long)R * C;
  if (i < n) {
    int r = (int)(i / C), c = (int)(i % C);
    y[(long)c * R + r] = (__bf16)x[i];
  }
}

// ---------------------------------------------------------------------------
// WMMA bf16 GEMM:  C[M,N] f32 = A[M,K] bf16 (row major) * BT[N,K] bf16 (B^T)
// block = 256 threads = 8 waves; block tile 128(M) x 64(N); wave tile 32x32.
// B tile staged to LDS (async global->LDS when available), double-buffered.
// Optional triple-store epilogue: C1 = C0, C2 = 2*C0 (output_virus / sum).
// ---------------------------------------------------------------------------
#define LDSROW 40  // 64x32 bf16 tile padded to 40 elems/row: conflict-free ds_load_b128

__global__ __launch_bounds__(256) void k_gemm_bf16_wmma(
    const __bf16* __restrict__ A, const __bf16* __restrict__ BT,
    float* __restrict__ C0, float* __restrict__ C1, float* __restrict__ C2,
    int M, int N, int K) {
  __shared__ __align__(16) __bf16 ldsB[2][64 * LDSROW];

  const int lane = threadIdx.x & 31;
  const int wave = threadIdx.x >> 5;
  const int wm = wave & 3;      // 4 waves along M
  const int wn = wave >> 2;     // 2 waves along N
  const int tm = blockIdx.y * 128 + wm * 32;
  const int tnb = blockIdx.x * 64;
  const int tn = tnb + wn * 32;
  const int l15 = lane & 15;
  const int hi = lane >> 4;
  const int aoff = hi * 8;      // A frag: lane half selects K 0..7 vs 8..15 (+16/ +24)
  const int boff = hi * 8;      // B frag: lane half selects K 0..7 vs 8..15 (+16/ +24)

  // B staging: 256 threads move 64 rows x 32 bf16 (one b128 per thread)
  const int srow = threadIdx.x >> 2;        // 0..63
  const int sseg = (threadIdx.x & 3) * 8;   // 0,8,16,24 (bf16 elems)
  int gn = tnb + srow; if (gn >= N) gn = N - 1;
  const __bf16* sbase = BT + (long)gn * K + sseg;

  int ra0 = tm + l15;      if (ra0 >= M) ra0 = M - 1;
  int ra1 = tm + 16 + l15; if (ra1 >= M) ra1 = M - 1;
  const __bf16* pa0 = A + (long)ra0 * K;
  const __bf16* pa1 = A + (long)ra1 * K;

  auto stage = [&](int buf, int k0) {
    __bf16* dst = &ldsB[buf][srow * LDSROW + sseg];
    const __bf16* src = sbase + k0;
#if HAVE_ASYNC_LDS
    __builtin_amdgcn_global_load_async_to_lds_b128(
        (as1_v4i*)(unsigned long long)src,
        (as3_v4i*)(unsigned int)(unsigned long long)dst, 0, 0);
#else
    *reinterpret_cast<u32x4*>(dst) = *reinterpret_cast<const u32x4*>(src);
#endif
  };

  floatx8 acc00 = {}, acc01 = {}, acc10 = {}, acc11 = {};
  const int nsteps = K >> 5;  // K % 32 == 0 for all shapes used here
  stage(0, 0);
  for (int i = 0; i < nsteps; i++) {
    const int k0 = i << 5;
    const int bb = i & 1;
    const bool more = (i + 1) < nsteps;
    if (more) {
      stage(bb ^ 1, k0 + 32);                       // overlap copy with math
      __builtin_prefetch(pa0 + k0 + 64, 0, 3);      // global_prefetch_b8
      __builtin_prefetch(pa1 + k0 + 64, 0, 3);
    }
#if HAVE_ASYNC_LDS
    if (more) wait_asynccnt_le<1>();                // current buffer landed
    else      wait_asynccnt_le<0>();
#endif
    __syncthreads();

    FragU a0, a1, b0, b1;
    a0.q[0] = *reinterpret_cast<const u32x4*>(pa0 + k0 + aoff);
    a0.q[1] = *reinterpret_cast<const u32x4*>(pa0 + k0 + aoff + 16);
    a1.q[0] = *reinterpret_cast<const u32x4*>(pa1 + k0 + aoff);
    a1.q[1] = *reinterpret_cast<const u32x4*>(pa1 + k0 + aoff + 16);
    const __bf16* lb0 = &ldsB[bb][(wn * 32 + l15) * LDSROW + boff];
    const __bf16* lb1 = lb0 + 16 * LDSROW;
    b0.q[0] = *reinterpret_cast<const u32x4*>(lb0);
    b0.q[1] = *reinterpret_cast<const u32x4*>(lb0 + 16);
    b1.q[0] = *reinterpret_cast<const u32x4*>(lb1);
    b1.q[1] = *reinterpret_cast<const u32x4*>(lb1 + 16);

    acc00 = __builtin_amdgcn_wmma_f32_16x16x32_bf16(false, a0.v, false, b0.v, (short)0, acc00, false, false);
    acc01 = __builtin_amdgcn_wmma_f32_16x16x32_bf16(false, a0.v, false, b1.v, (short)0, acc01, false, false);
    acc10 = __builtin_amdgcn_wmma_f32_16x16x32_bf16(false, a1.v, false, b0.v, (short)0, acc10, false, false);
    acc11 = __builtin_amdgcn_wmma_f32_16x16x32_bf16(false, a1.v, false, b1.v, (short)0, acc11, false, false);
    __syncthreads();   // readers done before this buffer is overwritten next step
  }

  const int mo = hi * 8;  // C/D layout: VGPR r holds M = r (+8 upper half), N = lane&15
  floatx8* accs[4] = { &acc00, &acc01, &acc10, &acc11 };
  for (int t = 0; t < 4; t++) {
    int ti = t >> 1, tj = t & 1;
    int n = tn + tj * 16 + l15;
    if (n >= N) continue;
    for (int r = 0; r < 8; r++) {
      int m = tm + ti * 16 + mo + r;
      if (m >= M) continue;
      long idx = (long)m * N + n;
      float v = (*accs[t])[r];
      C0[idx] = v;
      if (C1) { C1[idx] = v; C2[idx] = 2.0f * v; }
    }
  }
}

// ---------------------------------------------------------------------------
// small utility kernels
// ---------------------------------------------------------------------------
__global__ void k_fill(float* __restrict__ p, float v, long n) {
  long i = (long)blockIdx.x * blockDim.x + threadIdx.x;
  if (i < n) p[i] = v;
}

__global__ void k_feat_init(float* __restrict__ feat, const float* __restrict__ bias, long n) {
  long i = (long)blockIdx.x * blockDim.x + threadIdx.x;
  if (i < n) feat[i] = bias[i % HID];
}

__global__ void k_sum(const float* __restrict__ x, long n, float* __restrict__ out) {
  long i = (long)blockIdx.x * blockDim.x + threadIdx.x;
  long stride = (long)gridDim.x * blockDim.x;
  float s = 0.f;
  for (; i < n; i += stride) s += x[i];
  atomicAdd(out, s);
}

// edot[h] = sum_c lin_e[h*HID+c] * a_e[h*HID+c]
__global__ void k_edot(const float* __restrict__ lin_e, const float* __restrict__ a_e,
                       float* __restrict__ edot) {
  int h = threadIdx.x;
  if (h < HEADS) {
    float s = 0.f;
    for (int c = 0; c < HID; c++) s += lin_e[h * HID + c] * a_e[h * HID + c];
    edot[h] = s;
  }
}

// asrc[n,h], adst[n,h] from xw [N, HEADS*HID]
__global__ void k_attn_coef(const float* __restrict__ xw,
                            const float* __restrict__ a_src, const float* __restrict__ a_dst,
                            float* __restrict__ asrc, float* __restrict__ adst, int N) {
  long i = (long)blockIdx.x * blockDim.x + threadIdx.x;
  if (i >= (long)N * HEADS) return;
  int n = (int)(i / HEADS), h = (int)(i % HEADS);
  const float* xp = xw + (long)n * FDIM + h * HID;
  const float* sp = a_src + h * HID;
  const float* dp = a_dst + h * HID;
  float ss = 0.f, dd = 0.f;
  for (int c = 0; c < HID; c++) { float x = xp[c]; ss += x * sp[c]; dd += x * dp[c]; }
  asrc[i] = ss; adst[i] = dd;
}

__device__ __forceinline__ void atomicMaxF(float* addr, float val) {
  if (val >= 0.f) atomicMax((int*)addr, __float_as_int(val));
  else            atomicMin((unsigned int*)addr, __float_as_uint(val));
}

__device__ __forceinline__ void edge_sd(const int* __restrict__ ei, int E, int e,
                                        int& s, int& d) {
  if (e < E) { s = ei[e]; d = ei[E + e]; }
  else       { s = e - E; d = e - E; }  // appended self-loops
}

// pass 1: raw attention logits + leaky_relu(0.2) + segment max over dst
__global__ void k_alpha1(const float* __restrict__ asrc, const float* __restrict__ adst,
                         const int* __restrict__ ei, const float* __restrict__ ew,
                         const float* __restrict__ msum, float invE,
                         const float* __restrict__ edot,
                         float* __restrict__ alpha, float* __restrict__ segmax,
                         int E, int N) {
  long i = (long)blockIdx.x * blockDim.x + threadIdx.x;
  if (i >= (long)(E + N) * HEADS) return;
  int e = (int)(i / HEADS), h = (int)(i % HEADS);
  int s, d; edge_sd(ei, E, e, s, d);
  float a = asrc[(long)s * HEADS + h] + adst[(long)d * HEADS + h];
  if (ew) {
    float w = (e < E) ? ew[e] : msum[0] * invE;  // self-loop attr = mean(edge_weight)
    a += w * edot[h];
  }
  a = (a > 0.f) ? a : 0.2f * a;
  alpha[i] = a;
  atomicMaxF(&segmax[(long)d * HEADS + h], a);
}

// pass 2: exp(alpha - max) + segment sum
__global__ void k_alpha2(const int* __restrict__ ei, const float* __restrict__ segmax,
                         float* __restrict__ den, float* __restrict__ alpha, int E, int N) {
  long i = (long)blockIdx.x * blockDim.x + threadIdx.x;
  if (i >= (long)(E + N) * HEADS) return;
  int e = (int)(i / HEADS), h = (int)(i % HEADS);
  int s, d; edge_sd(ei, E, e, s, d); (void)s;
  float ex = expf(alpha[i] - segmax[(long)d * HEADS + h]);
  alpha[i] = ex;
  atomicAdd(&den[(long)d * HEADS + h], ex);
}

// pass 3: normalize
__global__ void k_alpha3(const int* __restrict__ ei, const float* __restrict__ den,
                         float* __restrict__ alpha, int E, int N) {
  long i = (long)blockIdx.x * blockDim.x + threadIdx.x;
  if (i >= (long)(E + N) * HEADS) return;
  int e = (int)(i / HEADS), h = (int)(i % HEADS);
  int s, d; edge_sd(ei, E, e, s, d); (void)s;
  alpha[i] = alpha[i] / (den[(long)d * HEADS + h] + 1e-16f);
}

// scatter: feat[dst] += alpha * xw[src] / HEADS, one thread per (e,h,c)
// -> 32 lanes hit consecutive addresses of one feature row (coalesced atomics)
__global__ void k_scatter(const float* __restrict__ xw, const float* __restrict__ alpha,
                          const int* __restrict__ ei, float* __restrict__ feat, int E, int N) {
  long i = (long)blockIdx.x * blockDim.x + threadIdx.x;
  if (i >= (long)(E + N) * FDIM) return;
  int e = (int)(i / FDIM);
  int r = (int)(i % FDIM);   // r = h*HID + c
  int h = r / HID;
  int s, d; edge_sd(ei, E, e, s, d);
  float a = alpha[(long)e * HEADS + h] * (1.0f / HEADS);
  atomicAdd(&feat[(long)d * HID + (r - h * HID)], a * xw[(long)s * FDIM + r]);
}

// BatchNorm (training, biased var): one block per column, LDS tree reduction
__global__ __launch_bounds__(256) void k_bn_stats(const float* __restrict__ y, int N,
                                                  float* __restrict__ mean,
                                                  float* __restrict__ istd) {
  __shared__ float ssum[256], ssq[256];
  int j = blockIdx.x;  // column 0..HID-1
  float s = 0.f, s2 = 0.f;
  for (int i = threadIdx.x; i < N; i += 256) {
    float v = y[(long)i * HID + j];
    s += v; s2 += v * v;
  }
  ssum[threadIdx.x] = s; ssq[threadIdx.x] = s2;
  __syncthreads();
  for (int w = 128; w > 0; w >>= 1) {
    if (threadIdx.x < w) {
      ssum[threadIdx.x] += ssum[threadIdx.x + w];
      ssq[threadIdx.x]  += ssq[threadIdx.x + w];
    }
    __syncthreads();
  }
  if (threadIdx.x == 0) {
    float m = ssum[0] / N;
    float var = ssq[0] / N - m * m;
    mean[j] = m;
    istd[j] = rsqrtf(var + 1e-5f);
  }
}

// BN apply + leaky_relu(0.01), in place, plus bf16 copy for the final WMMA GEMM
__global__ void k_bn_apply(float* __restrict__ y, const float* __restrict__ mean,
                           const float* __restrict__ istd, const float* __restrict__ g,
                           const float* __restrict__ b, __bf16* __restrict__ ybf, long n) {
  long i = (long)blockIdx.x * blockDim.x + threadIdx.x;
  if (i >= n) return;
  int j = (int)(i % HID);
  float v = (y[i] - mean[j]) * istd[j] * g[j] + b[j];
  v = (v > 0.f) ? v : 0.01f * v;
  y[i] = v;
  ybf[i] = (__bf16)v;
}

// ---------------------------------------------------------------------------
// host driver
// ---------------------------------------------------------------------------
extern "C" void kernel_launch(void* const* d_in, const int* in_sizes, int n_in,
                              void* d_out, int out_size, void* d_ws, size_t ws_size,
                              hipStream_t stream) {
  (void)in_sizes; (void)n_in; (void)out_size; (void)ws_size;

  const float* virus   = (const float*)d_in[0];
  const float* hostd   = (const float*)d_in[1];
  const float* coex    = (const float*)d_in[2];
  const float* v_ew    = (const float*)d_in[3];
  const float* h_ew    = (const float*)d_in[4];
  const float* vW      = (const float*)d_in[5];
  const float* v_asrc  = (const float*)d_in[6];
  const float* v_adst  = (const float*)d_in[7];
  const float* v_bias  = (const float*)d_in[8];
  const float* v_line  = (const float*)d_in[9];
  const float* v_atte  = (const float*)d_in[10];
  const float* hW      = (const float*)d_in[11];
  const float* h_asrc  = (const float*)d_in[12];
  const float* h_adst  = (const float*)d_in[13];
  const float* h_bias  = (const float*)d_in[14];
  const float* h_line  = (const float*)d_in[15];
  const float* h_atte  = (const float*)d_in[16];
  const float* vhW     = (const float*)d_in[17];
  const float* vh_asrc = (const float*)d_in[18];
  const float* vh_adst = (const float*)d_in[19];
  const float* vh_bias = (const float*)d_in[20];
  const float* hvW     = (const float*)d_in[21];
  const float* hv_asrc = (const float*)d_in[22];
  const float* hv_adst = (const float*)d_in[23];
  const float* hv_bias = (const float*)d_in[24];
  const float* linv_W  = (const float*)d_in[25];
  const float* linh_W  = (const float*)d_in[27];
  const float* bn_g    = (const float*)d_in[29];
  const float* bn_b    = (const float*)d_in[30];
  const int*   vei     = (const int*)d_in[31];
  const int*   hei     = (const int*)d_in[32];
  const int*   cei     = (const int*)d_in[33];
  const int*   cei_t   = (const int*)d_in[34];
  // linv_b/linh_b (26/28) cancel exactly under BatchNorm -> not needed.

  float* out = (float*)d_out;
  const long OUT_HOST  = 0;
  const long OUT_VIRUS = (long)NVN * NHN;
  const long OUT_SUM   = 2L * NVN * NHN;
  const long A_V  = 3L * NVN * NHN;
  const long A_H  = A_V  + (long)(EVE + NVN) * HEADS;
  const long A_VH = A_H  + (long)(EHE + NHN) * HEADS;
  const long A_HV = A_VH + (long)(ECE + DVD) * HEADS;
  const long F_V  = A_HV + (long)(ECE + DHD) * HEADS;
  const long F_H  = F_V  + (long)NVN * HID;
  const long F_VH = F_H  + (long)NHN * HID;
  const long F_HV = F_VH + (long)DVD * HID;

  // bump allocator over workspace (256B aligned for b128 loads)
  char* wsb = (char*)d_ws;
  size_t woff = 0;
  auto alloc = [&](size_t bytes) -> void* {
    void* p = wsb + woff;
    woff = (woff + bytes + 255) & ~(size_t)255;
    return p;
  };
  auto cdiv = [](long a, long b) { return (unsigned int)((a + b - 1) / b); };

  __bf16* v_bf   = (__bf16*)alloc((size_t)NVN * DVD * 2);
  __bf16* h_bf   = (__bf16*)alloc((size_t)NHN * DHD * 2);
  __bf16* co_bf  = (__bf16*)alloc((size_t)DVD * DHD * 2);
  __bf16* coT_bf = (__bf16*)alloc((size_t)DVD * DHD * 2);
  __bf16* vWT    = (__bf16*)alloc((size_t)FDIM * DVD * 2);
  __bf16* hWT    = (__bf16*)alloc((size_t)FDIM * DHD * 2);
  __bf16* vhWT   = (__bf16*)alloc((size_t)FDIM * DHD * 2);
  __bf16* hvWT   = (__bf16*)alloc((size_t)FDIM * DVD * 2);
  __bf16* lvWT   = (__bf16*)alloc((size_t)HID * DVD * 2);
  __bf16* lhWT   = (__bf16*)alloc((size_t)HID * DHD * 2);
  float*  xw_v   = (float*)alloc((size_t)NVN * FDIM * 4);
  float*  xw_h   = (float*)alloc((size_t)NHN * FDIM * 4);
  float*  xw_vh  = (float*)alloc((size_t)DVD * FDIM * 4);
  float*  xw_hv  = (float*)alloc((size_t)DHD * FDIM * 4);
  float*  y_v    = (float*)alloc((size_t)NVN * HID * 4);
  float*  y_h    = (float*)alloc((size_t)NHN * HID * 4);
  __bf16* vh_bf  = (__bf16*)alloc((size_t)NVN * HID * 2);
  __bf16* hh_bf  = (__bf16*)alloc((size_t)NHN * HID * 2);
  float*  bn_m   = (float*)alloc(2 * HID * 4);
  float*  bn_i   = (float*)alloc(2 * HID * 4);

  // --- precision conversion (activations row-major bf16; weights transposed bf16) ---
  k_cvt <<<cdiv((long)NVN * DVD, 256), 256, 0, stream>>>(virus, v_bf, (long)NVN * DVD);
  k_cvt <<<cdiv((long)NHN * DHD, 256), 256, 0, stream>>>(hostd, h_bf, (long)NHN * DHD);
  k_cvt <<<cdiv((long)DVD * DHD, 256), 256, 0, stream>>>(coex, co_bf, (long)DVD * DHD);
  k_cvtT<<<cdiv((long)DVD * DHD, 256), 256, 0, stream>>>(coex, coT_bf, DVD, DHD);
  k_cvtT<<<cdiv((long)DVD * FDIM, 256), 256, 0, stream>>>(vW,   vWT,  DVD, FDIM);
  k_cvtT<<<cdiv((long)DHD * FDIM, 256), 256, 0, stream>>>(hW,   hWT,  DHD, FDIM);
  k_cvtT<<<cdiv((long)DHD * FDIM, 256), 256, 0, stream>>>(vhW,  vhWT, DHD, FDIM);
  k_cvtT<<<cdiv((long)DVD * FDIM, 256), 256, 0, stream>>>(hvW,  hvWT, DVD, FDIM);
  k_cvtT<<<cdiv((long)DVD * HID,  256), 256, 0, stream>>>(linv_W, lvWT, DVD, HID);
  k_cvtT<<<cdiv((long)DHD * HID,  256), 256, 0, stream>>>(linh_W, lhWT, DHD, HID);

  auto gemm = [&](const __bf16* A, const __bf16* BT, float* C0, float* C1, float* C2,
                  int M, int N, int K) {
    dim3 g((N + 63) / 64, (M + 127) / 128);
    k_gemm_bf16_wmma<<<g, 256, 0, stream>>>(A, BT, C0, C1, C2, M, N, K);
  };

  // --- feature projections (WMMA) ---
  gemm(v_bf,   vWT,  xw_v,  nullptr, nullptr, NVN, FDIM, DVD);
  gemm(h_bf,   hWT,  xw_h,  nullptr, nullptr, NHN, FDIM, DHD);
  gemm(co_bf,  vhWT, xw_vh, nullptr, nullptr, DVD, FDIM, DHD);
  gemm(coT_bf, hvWT, xw_hv, nullptr, nullptr, DHD, FDIM, DVD);
  gemm(v_bf,   lvWT, y_v,   nullptr, nullptr, NVN, HID, DVD);
  gemm(h_bf,   lhWT, y_h,   nullptr, nullptr, NHN, HID, DHD);

  // --- GAT layers ---
  auto run_gat = [&](const float* xw, int N, const int* ei, int E,
                     const float* a_src, const float* a_dst, const float* bias,
                     const float* ew, const float* lin_e, const float* a_e,
                     float* alpha_out, float* feat_out) {
    float* asrc   = (float*)alloc((size_t)N * HEADS * 4);
    float* adst   = (float*)alloc((size_t)N * HEADS * 4);
    float* segmax = (float*)alloc((size_t)N * HEADS * 4);
    float* den    = (float*)alloc((size_t)N * HEADS * 4);
    float* msum   = (float*)alloc(256);
    float* edot   = (float*)alloc(256);
    long nh = (long)N * HEADS;
    long tot = (long)(E + N) * HEADS;

    k_attn_coef<<<cdiv(nh, 256), 256, 0, stream>>>(xw, a_src, a_dst, asrc, adst, N);
    k_fill<<<cdiv(nh, 256), 256, 0, stream>>>(segmax, -INFINITY, nh);
    k_fill<<<cdiv(nh, 256), 256, 0, stream>>>(den, 0.f, nh);
    k_feat_init<<<cdiv((long)N * HID, 256), 256, 0, stream>>>(feat_out, bias, (long)N * HID);
    if (ew) {
      k_fill<<<1, 256, 0, stream>>>(msum, 0.f, 1);
      k_sum<<<64, 256, 0, stream>>>(ew, (long)E, msum);
      k_edot<<<1, 32, 0, stream>>>(lin_e, a_e, edot);
    }
    k_alpha1<<<cdiv(tot, 256), 256, 0, stream>>>(asrc, adst, ei, ew, msum, 1.0f / E, edot,
                                                 alpha_out, segmax, E, N);
    k_alpha2<<<cdiv(tot, 256), 256, 0, stream>>>(ei, segmax, den, alpha_out, E, N);
    k_alpha3<<<cdiv(tot, 256), 256, 0, stream>>>(ei, den, alpha_out, E, N);
    k_scatter<<<cdiv((long)(E + N) * FDIM, 256), 256, 0, stream>>>(xw, alpha_out, ei,
                                                                   feat_out, E, N);
  };

  run_gat(xw_v,  NVN, vei,   EVE, v_asrc,  v_adst,  v_bias,  v_ew,    v_line,  v_atte,
          out + A_V,  out + F_V);
  run_gat(xw_h,  NHN, hei,   EHE, h_asrc,  h_adst,  h_bias,  h_ew,    h_line,  h_atte,
          out + A_H,  out + F_H);
  run_gat(xw_vh, DVD, cei_t, ECE, vh_asrc, vh_adst, vh_bias, nullptr, nullptr, nullptr,
          out + A_VH, out + F_VH);
  run_gat(xw_hv, DHD, cei,   ECE, hv_asrc, hv_adst, hv_bias, nullptr, nullptr, nullptr,
          out + A_HV, out + F_HV);

  // --- BatchNorm + leaky_relu(0.01) (linear bias cancels under BN) ---
  k_bn_stats<<<HID, 256, 0, stream>>>(y_v, NVN, bn_m, bn_i);
  k_bn_stats<<<HID, 256, 0, stream>>>(y_h, NHN, bn_m + HID, bn_i + HID);
  k_bn_apply<<<cdiv((long)NVN * HID, 256), 256, 0, stream>>>(y_v, bn_m, bn_i, bn_g, bn_b,
                                                             vh_bf, (long)NVN * HID);
  k_bn_apply<<<cdiv((long)NHN * HID, 256), 256, 0, stream>>>(y_h, bn_m + HID, bn_i + HID,
                                                             bn_g, bn_b, hh_bf, (long)NHN * HID);

  // --- final score GEMM: G = vh @ hh.T; output_virus == G; sum == 2G ---
  gemm(vh_bf, hh_bf, out + OUT_HOST, out + OUT_VIRUS, out + OUT_SUM, NVN, NHN, HID);
}
